// DeformNet_49495203119120
// MI455X (gfx1250) — matrix operand order repaired and checked by reading, not compiled, not run on hardware
//
#include <hip/hip_runtime.h>
#include <hip/hip_bf16.h>
#include <math.h>
#include <stdint.h>

typedef __attribute__((ext_vector_type(16))) __bf16 v16bf;
typedef __attribute__((ext_vector_type(8)))  __bf16 v8bf;
typedef __attribute__((ext_vector_type(8)))  float  v8f;

#define BB 2
#define HH 448
#define WW 640
#define HC 112
#define WC 160
#define CP 32          // channel padding for 28-ch stage
#define CPF 576        // 565 padded to 18*32

static __device__ __forceinline__ int clampi(int v, int lo, int hi) {
    return v < lo ? lo : (v > hi ? hi : v);
}

// 16-bit A-matrix K index for element e, lane-half hi: contiguous k = 16*hi + e (matches B loads)
static __device__ __forceinline__ int kAof(int e, int hi) {
    return e + (hi ? 16 : 0);
}

// ---------------------------------------------------------------------------
__global__ void k_fill0(uint32_t* __restrict__ p, long n) {
    long i = (long)blockIdx.x * blockDim.x + threadIdx.x;
    if (i < n) p[i] = 0u;
}

// features2 NCHW f32 -> NHWC padded bf16 (Hc+2, Wc+2, 576)
__global__ void k_cvt_feat(const float* __restrict__ s, __bf16* __restrict__ d) {
    long gid = (long)blockIdx.x * blockDim.x + threadIdx.x;
    if (gid >= (long)BB * 565 * HC * WC) return;
    int x = gid % WC; long r = gid / WC;
    int y = r % HC; r /= HC;
    int c = r % 565; int b = r / 565;
    d[(((long)b * (HC + 2) + (y + 1)) * (WC + 2) + (x + 1)) * CPF + c] =
        (__bf16)s[gid];
}

// pack transpose-conv weights (Cin,Cout,4,4) -> per-lane A fragments
// layout: [py*2+pxp][octile][tt][icc][lane][e]
__global__ void k_pack_wT(const float* __restrict__ w, __bf16* __restrict__ out,
                          int Cin, int Cout, int nCh, int total) {
    int gid = blockIdx.x * blockDim.x + threadIdx.x;
    if (gid >= total) return;
    int e = gid & 15;
    int lane = (gid >> 4) & 31;
    int rest = gid >> 9;
    int icc = rest % nCh; rest /= nCh;
    int tt = rest & 3; rest >>= 2;
    int ocT = Cout >> 4;
    int octile = rest % ocT; rest /= ocT;
    int pxp = rest & 1; int py = rest >> 1;
    int hi = lane >> 4, nl = lane & 15;
    int ka = icc * 32 + kAof(e, hi);
    int oc = octile * 16 + nl;
    int ty = tt >> 1, tx = tt & 1;
    int wky = (py == 0) ? (ty == 0 ? 3 : 1) : (ty == 0 ? 2 : 0);
    int wkx = (pxp == 0) ? (tx == 0 ? 3 : 1) : (tx == 0 ? 2 : 0);
    out[gid] = (ka < Cin) ? (__bf16)w[((ka * Cout + oc) * 4 + wky) * 4 + wkx]
                          : (__bf16)0.0f;
}

// pack 3x3 conv weights (16,Cin,3,3), K=32 per tap -> [tt(9)][lane][e]
__global__ void k_pack_w3(const float* __restrict__ w, __bf16* __restrict__ out, int Cin) {
    int gid = blockIdx.x * blockDim.x + threadIdx.x;
    if (gid >= 9 * 32 * 16) return;
    int e = gid & 15;
    int lane = (gid >> 4) & 31;
    int tt = gid >> 9;
    int hi = lane >> 4, nl = lane & 15;
    int ka = kAof(e, hi);
    out[gid] = (ka < Cin) ? (__bf16)w[(nl * Cin + ka) * 9 + tt] : (__bf16)0.0f;
}

// pack 3x3 conv weights (16,16,3,3), tap-PAIRED K=32 (two kx per chunk)
// chunks cc: ky=cc>>1; cc&1==0 -> kx 0,1 ; cc&1==1 -> kx 2 (upper K zero)
__global__ void k_pack_w3p(const float* __restrict__ w, __bf16* __restrict__ out) {
    int gid = blockIdx.x * blockDim.x + threadIdx.x;
    if (gid >= 6 * 32 * 16) return;
    int e = gid & 15;
    int lane = (gid >> 4) & 31;
    int cc = gid >> 9;
    int hi = lane >> 4, nl = lane & 15;
    int ka = kAof(e, hi);
    int tap = ka >> 4, ic = ka & 15;
    int ky = cc >> 1, half = cc & 1;
    int kx = (half == 0) ? tap : (tap == 0 ? 2 : -1);
    out[gid] = (kx >= 0) ? (__bf16)w[(nl * 16 + ic) * 9 + ky * 3 + kx] : (__bf16)0.0f;
}

// ---------------------------------------------------------------------------
// flow upsample + grid_sample + validity; writes mask_input ch 16..27 (NHWC)
// ---------------------------------------------------------------------------
__global__ void k_warp(const float* __restrict__ x1, const float* __restrict__ x2,
                       const float* __restrict__ flow2, const int* __restrict__ anchors,
                       float* __restrict__ out_tm, float* __restrict__ out_valid,
                       __bf16* __restrict__ hin /* (B,H+2,W+2,32) */) {
    int gid = blockIdx.x * blockDim.x + threadIdx.x;
    if (gid >= BB * HH * WW) return;
    int w = gid % WW; int r = gid / WW; int h = r % HH; int b = r / HH;

    float sy = (h + 0.5f) * 0.25f - 0.5f;
    float sx = (w + 0.5f) * 0.25f - 0.5f;
    int y0 = (int)floorf(sy); float fy = sy - (float)y0;
    int x0 = (int)floorf(sx); float fx = sx - (float)x0;
    int y0c = clampi(y0, 0, HC - 1), y1c = clampi(y0 + 1, 0, HC - 1);
    int x0c = clampi(x0, 0, WC - 1), x1c = clampi(x0 + 1, 0, WC - 1);
    const float* fp = flow2 + (size_t)b * 2 * HC * WC;
    float flo[2];
#pragma unroll
    for (int c = 0; c < 2; ++c) {
        const float* pl = fp + c * HC * WC;
        float v00 = pl[y0c * WC + x0c], v01 = pl[y0c * WC + x1c];
        float v10 = pl[y1c * WC + x0c], v11 = pl[y1c * WC + x1c];
        flo[c] = 20.0f * ((v00 * (1.f - fx) + v01 * fx) * (1.f - fy)
                        + (v10 * (1.f - fx) + v11 * fx) * fy);
    }
    float xw = (float)w + flo[0], yw = (float)h + flo[1];
    float gx = xw / (float)(WW - 1) * 2.f - 1.f;
    float gy = yw / (float)(HH - 1) * 2.f - 1.f;
    float px = (gx + 1.f) * 0.5f * (float)WW - 0.5f;
    float py = (gy + 1.f) * 0.5f * (float)HH - 0.5f;
    int ix0 = (int)floorf(px), iy0 = (int)floorf(py);
    float wx = px - (float)ix0, wy = py - (float)iy0;

    const float* x2b = x2 + (size_t)b * 6 * HH * WW;
    const float* x1b = x1 + (size_t)b * 6 * HH * WW;

    float wt[4] = {(1.f - wx) * (1.f - wy), wx * (1.f - wy),
                   (1.f - wx) * wy, wx * wy};
    int xs[4] = {ix0, ix0 + 1, ix0, ix0 + 1};
    int ys[4] = {iy0, iy0, iy0 + 1, iy0 + 1};

    float tm[3] = {0.f, 0.f, 0.f}, rgb[3] = {0.f, 0.f, 0.f}, tmv = 0.f;
#pragma unroll
    for (int q = 0; q < 4; ++q) {
        int ix = xs[q], iy = ys[q];
        if ((ix >= 0) & (ix < WW) & (iy >= 0) & (iy < HH)) {
            int base = iy * WW + ix;
#pragma unroll
            for (int c = 0; c < 3; ++c) {
                rgb[c] += wt[q] * x2b[(size_t)c * HH * WW + base];
                tm[c]  += wt[q] * x2b[(size_t)(c + 3) * HH * WW + base];
            }
            float z = x2b[(size_t)5 * HH * WW + base];
            tmv += wt[q] * ((z > 0.f && z <= 6.0f) ? 1.f : 0.f);
        }
    }
    float srcz = x1b[(size_t)5 * HH * WW + h * WW + w];
    const int* an = anchors + ((size_t)(b * HH + h) * WW + w) * 4;
    bool anc_ok = (an[0] >= 0) & (an[1] >= 0) & (an[2] >= 0) & (an[3] >= 0);
    bool v_src = (srcz > 0.f) && (srcz <= 6.0f) && anc_ok;
    bool v_tgt = (tm[2] > 0.f) && (tm[2] <= 6.0f) && (tmv >= 0.999f);

    int pix = h * WW + w;
#pragma unroll
    for (int c = 0; c < 3; ++c)
        out_tm[((size_t)(b * 3 + c) * HH) * WW + pix] = tm[c];
    out_valid[(size_t)b * HH * WW + pix] = (v_src && v_tgt) ? 1.0f : 0.0f;

    __bf16* hp = hin + (((size_t)b * (HH + 2) + (h + 1)) * (WW + 2) + (w + 1)) * CP;
#pragma unroll
    for (int c = 0; c < 6; ++c)
        hp[16 + c] = (__bf16)x1b[(size_t)c * HH * WW + pix];
#pragma unroll
    for (int c = 0; c < 3; ++c) {
        hp[22 + c] = (__bf16)rgb[c];
        hp[25 + c] = (__bf16)tm[c];
    }
}

// ---------------------------------------------------------------------------
// 4x4 stride-2 transpose conv, implicit GEMM on v_wmma_f32_16x16x32_bf16.
// in: NHWC padded (Hin+2,Win+2,Cpad). A pre-packed per parity.
// out: NHWC padded (Hout+2,Wout+2,CoutTot), writes 16 oc per wave.
// ---------------------------------------------------------------------------
__global__ void k_convT4x4_wmma(const __bf16* __restrict__ in, const __bf16* __restrict__ wpack,
                                const float* __restrict__ bias, __bf16* __restrict__ outb,
                                int Cpad, int nCh, int Cout, int CoutTot,
                                int Hin, int Win, int Hout, int Wout, int nWaves) {
    int gid = blockIdx.x * blockDim.x + threadIdx.x;
    int wid = gid >> 5;
    int lane = threadIdx.x & 31;
    if (wid >= nWaves) return;

    int tilesX = Wout >> 4;
    int tx2 = wid % tilesX; int r = wid / tilesX;
    int oy = r % Hout; r /= Hout;
    int ocT = Cout >> 4;
    int octile = r % ocT; int b = r / ocT;

    int pxp = tx2 & 1;
    int oxbase = (tx2 >> 1) * 32 + pxp;
    int nl = lane & 15, hi = lane >> 4;
    int myox = oxbase + 2 * nl;
    int py = oy & 1;

    int iyT[2], ixT[2];
    if (py == 0) { iyT[0] = ((oy - 2) >> 1) + 1; iyT[1] = (oy >> 1) + 1; }
    else         { iyT[0] = ((oy - 1) >> 1) + 1; iyT[1] = ((oy + 1) >> 1) + 1; }
    if (pxp == 0){ ixT[0] = ((myox - 2) >> 1) + 1; ixT[1] = (myox >> 1) + 1; }
    else         { ixT[0] = ((myox - 1) >> 1) + 1; ixT[1] = ((myox + 1) >> 1) + 1; }

    int kB0 = hi ? 16 : 0;
    const __bf16* inb = in + (size_t)b * (Hin + 2) * (Win + 2) * Cpad;
    const __bf16* wp = wpack + ((((size_t)(py * 2 + pxp) * ocT + octile) * 4) * nCh) * 512
                             + (size_t)lane * 16;

    v8f acc = {};
#pragma unroll
    for (int tt = 0; tt < 4; ++tt) {
        const __bf16* brow = inb + ((size_t)iyT[tt >> 1] * (Win + 2) + ixT[tt & 1]) * Cpad + kB0;
        const __bf16* arow = wp + (size_t)tt * nCh * 512;
        __builtin_prefetch(brow, 0, 3);
        for (int icc = 0; icc < nCh; ++icc) {
            v16bf av = *(const v16bf*)(arow + (size_t)icc * 512);
            v16bf bv = *(const v16bf*)(brow + icc * 32);
            acc = __builtin_amdgcn_wmma_f32_16x16x32_bf16(false, av, false, bv,
                                                          (short)0, acc, false, false);
        }
    }
    v8bf ov;
#pragma unroll
    for (int v = 0; v < 8; ++v) {
        int oc = octile * 16 + v + (hi ? 8 : 0);
        ov[v] = (__bf16)(acc[v] + bias[oc]);
    }
    size_t o = (((size_t)b * (Hout + 2) + (oy + 1)) * (Wout + 2) + (myox + 1)) * CoutTot
             + octile * 16 + (hi ? 8 : 0);
    *(v8bf*)&outb[o] = ov;
}

// ---------------------------------------------------------------------------
// 3x3 conv, 28->16 (c0): input NHWC CP=32, K=32 per tap (9 WMMAs).
// Output: f32 NCHW + bf16 NHWC C=16.
// ---------------------------------------------------------------------------
__global__ void k_conv3x3_c32(const __bf16* __restrict__ in, const __bf16* __restrict__ wpack,
                              const float* __restrict__ bias,
                              float* __restrict__ outf, __bf16* __restrict__ outb,
                              int nWaves) {
    int gid = blockIdx.x * blockDim.x + threadIdx.x;
    int wid = gid >> 5;
    int lane = threadIdx.x & 31;
    if (wid >= nWaves) return;

    int tilesX = WW >> 4;
    int tx = wid % tilesX; int r = wid / tilesX;
    int oy = r % HH; int b = r / HH;
    int ox0 = tx << 4;
    int nl = lane & 15, hi = lane >> 4;
    int myx = ox0 + nl;
    int kB0 = hi ? 16 : 0;
    const int Wp = WW + 2;
    const __bf16* inb = in + (size_t)b * (HH + 2) * Wp * CP + kB0;
    const __bf16* wp = wpack + (size_t)lane * 16;

    v8f acc = {};
#pragma unroll
    for (int tt = 0; tt < 9; ++tt) {
        int ky = tt / 3, kx = tt - ky * 3;
        v16bf av = *(const v16bf*)(wp + (size_t)tt * 512);
        v16bf bv = *(const v16bf*)(inb + ((size_t)(oy + ky) * Wp + (myx + kx)) * CP);
        acc = __builtin_amdgcn_wmma_f32_16x16x32_bf16(false, av, false, bv,
                                                      (short)0, acc, false, false);
    }
    v8bf ov;
#pragma unroll
    for (int v = 0; v < 8; ++v) {
        int oc = v + (hi ? 8 : 0);
        float val = acc[v] + bias[oc];
        val = val > 0.f ? val : 0.f;                       // relu
        outf[((size_t)(b * 16 + oc) * HH + oy) * WW + myx] = val;
        ov[v] = (__bf16)val;
    }
    size_t o = (((size_t)b * (HH + 2) + (oy + 1)) * Wp + (myx + 1)) * 16 + (hi ? 8 : 0);
    *(v8bf*)&outb[o] = ov;
}

// ---------------------------------------------------------------------------
// 3x3 conv 16->16: input NHWC C=16; two horizontal taps packed per K=32 WMMA
// (6 WMMAs). Optional residual (f32 NCHW) + ReLU; writes f32 and/or bf16 C=16.
// ---------------------------------------------------------------------------
__global__ void k_conv3x3_c16(const __bf16* __restrict__ in, const __bf16* __restrict__ wpack,
                              const float* __restrict__ bias, const float* __restrict__ resid,
                              float* __restrict__ outf, __bf16* __restrict__ outb,
                              int nWaves) {
    int gid = blockIdx.x * blockDim.x + threadIdx.x;
    int wid = gid >> 5;
    int lane = threadIdx.x & 31;
    if (wid >= nWaves) return;

    int tilesX = WW >> 4;
    int tx = wid % tilesX; int r = wid / tilesX;
    int oy = r % HH; int b = r / HH;
    int ox0 = tx << 4;
    int nl = lane & 15, hi = lane >> 4;
    int myx = ox0 + nl;
    int kB0 = hi ? 16 : 0;
    const int Wp = WW + 2;
    const __bf16* inb = in + (size_t)b * (HH + 2) * Wp * 16 + kB0;
    const __bf16* wp = wpack + (size_t)lane * 16;

    v8f acc = {};
#pragma unroll
    for (int cc = 0; cc < 6; ++cc) {
        int ky = cc >> 1, kxb = (cc & 1) * 2;   // pixel pair (kxb, kxb+1) contiguous
        v16bf av = *(const v16bf*)(wp + (size_t)cc * 512);
        v16bf bv = *(const v16bf*)(inb + ((size_t)(oy + ky) * Wp + (myx + kxb)) * 16);
        acc = __builtin_amdgcn_wmma_f32_16x16x32_bf16(false, av, false, bv,
                                                      (short)0, acc, false, false);
    }
    v8bf ov;
#pragma unroll
    for (int v = 0; v < 8; ++v) {
        int oc = v + (hi ? 8 : 0);
        float val = acc[v] + bias[oc];
        size_t idx = ((size_t)(b * 16 + oc) * HH + oy) * WW + myx;
        if (resid) val += resid[idx];
        val = val > 0.f ? val : 0.f;                       // relu
        if (outf) outf[idx] = val;
        ov[v] = (__bf16)val;
    }
    if (outb) {
        size_t o = (((size_t)b * (HH + 2) + (oy + 1)) * Wp + (myx + 1)) * 16 + (hi ? 8 : 0);
        *(v8bf*)&outb[o] = ov;
    }
}

// ---------------------------------------------------------------------------
__global__ void k_outconv(const float* __restrict__ hbuf, const float* __restrict__ w1,
                          const float* __restrict__ b1, float* __restrict__ out_mask,
                          float* __restrict__ out_cw) {
    int gid = blockIdx.x * blockDim.x + threadIdx.x;
    if (gid >= BB * HH * WW) return;
    int x = gid % WW; int r = gid / WW; int y = r % HH; int b = r / HH;
    float acc = b1[0];
#pragma unroll
    for (int ic = 0; ic < 16; ++ic) {
        const float* pl = hbuf + (size_t)(b * 16 + ic) * HH * WW;
#pragma unroll
        for (int ky = 0; ky < 3; ++ky) {
            int iy = y + ky - 1;
            if (iy < 0 || iy >= HH) continue;
#pragma unroll
            for (int kx = 0; kx < 3; ++kx) {
                int ix = x + kx - 1;
                if (ix < 0 || ix >= WW) continue;
                acc += pl[iy * WW + ix] * w1[(ic * 3 + ky) * 3 + kx];
            }
        }
    }
    float s = 1.f / (1.f + expf(-acc));
    out_mask[gid] = s;
    out_cw[gid] = s;
}

// ---------------------------------------------------------------------------
extern "C" void kernel_launch(void* const* d_in, const int* in_sizes, int n_in,
                              void* d_out, int out_size, void* d_ws, size_t ws_size,
                              hipStream_t stream) {
    (void)in_sizes; (void)n_in; (void)out_size; (void)ws_size;
    const float* x1     = (const float*)d_in[0];
    const float* x2     = (const float*)d_in[1];
    const float* flow2  = (const float*)d_in[2];
    const float* feats  = (const float*)d_in[3];
    const int*   anchrs = (const int*)d_in[4];
    const float* c0_b  = (const float*)d_in[11];
    const float* c0_w  = (const float*)d_in[12];
    const float* out_b = (const float*)d_in[13];
    const float* out_w = (const float*)d_in[14];
    const float* rW[6] = { (const float*)d_in[16], (const float*)d_in[18],
                           (const float*)d_in[20], (const float*)d_in[22],
                           (const float*)d_in[24], (const float*)d_in[26] };
    const float* rB[6] = { (const float*)d_in[15], (const float*)d_in[17],
                           (const float*)d_in[19], (const float*)d_in[21],
                           (const float*)d_in[23], (const float*)d_in[25] };
    const float* up1_b = (const float*)d_in[27];
    const float* up1_w = (const float*)d_in[28];
    const float* up2_b = (const float*)d_in[29];
    const float* up2_w = (const float*)d_in[30];

    float* o_mask  = (float*)d_out;
    float* o_tm    = o_mask + (size_t)BB * HH * WW;
    float* o_valid = o_tm   + (size_t)BB * 3 * HH * WW;
    float* o_cw    = o_valid + (size_t)BB * HH * WW;

    char* ws = (char*)d_ws;
    size_t off = 0;
    auto carve = [&](size_t elems, size_t esz) -> void* {
        void* p = ws + off;
        off += (elems * esz + 255) & ~(size_t)255;
        return p;
    };
    const size_t nFeatP  = (size_t)BB * (HC + 2) * (WC + 2) * CPF;
    const size_t nF1P    = (size_t)BB * 226 * 322 * CP;
    const size_t nHinP   = (size_t)BB * (HH + 2) * (WW + 2) * CP;
    const size_t nAct16  = (size_t)BB * (HH + 2) * (WW + 2) * 16 + 64; // +tail pad
    __bf16* featP = (__bf16*)carve(nFeatP, 2);
    __bf16* f1P   = (__bf16*)carve(nF1P, 2);
    __bf16* hinP  = (__bf16*)carve(nHinP, 2);
    __bf16* hP    = (__bf16*)carve(nAct16, 2);
    __bf16* tP    = (__bf16*)carve(nAct16, 2);
    float*  hF    = (float*)carve((size_t)BB * 16 * HH * WW, 4);
    const int nWT1 = 4 * 2 * 4 * 18 * 512;   // 294912
    const int nWT2 = 4 * 1 * 4 * 1 * 512;    // 8192
    __bf16* wpUp1 = (__bf16*)carve(nWT1, 2);
    __bf16* wpUp2 = (__bf16*)carve(nWT2, 2);
    __bf16* wpC0  = (__bf16*)carve(9 * 512, 2);
    __bf16* wpR[6];
    for (int i = 0; i < 6; ++i) wpR[i] = (__bf16*)carve(6 * 512, 2);

    auto fill = [&](void* p, size_t nbf16) {
        long n32 = (long)(nbf16 / 2);
        k_fill0<<<(unsigned)((n32 + 255) / 256), 256, 0, stream>>>((uint32_t*)p, n32);
    };
    fill(featP, nFeatP);
    fill(f1P, nF1P);
    fill(hinP, nHinP);
    fill(hP, nAct16);
    fill(tP, nAct16);

    {
        long n = (long)BB * 565 * HC * WC;
        k_cvt_feat<<<(unsigned)((n + 255) / 256), 256, 0, stream>>>(feats, featP);
    }
    k_pack_wT<<<(nWT1 + 255) / 256, 256, 0, stream>>>(up1_w, wpUp1, 565, 32, 18, nWT1);
    k_pack_wT<<<(nWT2 + 255) / 256, 256, 0, stream>>>(up2_w, wpUp2, 32, 16, 1, nWT2);
    k_pack_w3<<<(9 * 512 + 255) / 256, 256, 0, stream>>>(c0_w, wpC0, 28);
    for (int i = 0; i < 6; ++i)
        k_pack_w3p<<<(6 * 512 + 255) / 256, 256, 0, stream>>>(rW[i], wpR[i]);

    k_warp<<<(BB * HH * WW + 255) / 256, 256, 0, stream>>>(x1, x2, flow2, anchrs,
                                                           o_tm, o_valid, hinP);
    {   // up1: 565->32, 112x160 -> 224x320
        int nW = BB * 2 * 224 * (320 >> 4); // 17920
        k_convT4x4_wmma<<<nW / 8, 256, 0, stream>>>(featP, wpUp1, up1_b, f1P,
                                                    CPF, 18, 32, CP, HC, WC, 224, 320, nW);
    }
    {   // up2: 32->16, 224x320 -> 448x640 into hinP ch0..15
        int nW = BB * 1 * HH * (WW >> 4); // 35840
        k_convT4x4_wmma<<<nW / 8, 256, 0, stream>>>(f1P, wpUp2, up2_b, hinP,
                                                    CP, 1, 16, CP, 224, 320, HH, WW, nW);
    }
    int nW3 = BB * HH * (WW >> 4); // 35840
    k_conv3x3_c32<<<nW3 / 8, 256, 0, stream>>>(hinP, wpC0, c0_b, hF, hP, nW3);
    for (int rr = 0; rr < 3; ++rr) {
        k_conv3x3_c16<<<nW3 / 8, 256, 0, stream>>>(hP, wpR[2 * rr], rB[2 * rr],
                                                   nullptr, nullptr, tP, nW3);
        k_conv3x3_c16<<<nW3 / 8, 256, 0, stream>>>(tP, wpR[2 * rr + 1], rB[2 * rr + 1],
                                                   hF, hF, hP, nW3);
    }
    k_outconv<<<(BB * HH * WW + 255) / 256, 256, 0, stream>>>(hF, out_w, out_b,
                                                              o_mask, o_cw);
}
